// AtnPool_28200755265840
// MI455X (gfx1250) — matrix operand order repaired and checked by reading, not compiled, not run on hardware
//
#include <hip/hip_runtime.h>
#include <hip/hip_bf16.h>
#include <stdint.h>

// Problem constants
#define BATCH 32
#define SEQ   2048
#define DIM   1024
#define NHEAD 8
#define DHEAD 64     // hidden per head (DH)
#define DHOUT 128    // score width per head (DHO)

typedef __attribute__((ext_vector_type(16))) __bf16 v16bf;
typedef __attribute__((ext_vector_type(8)))  float  v8f;

// fp32 -> bf16 pair, round-to-nearest (ties away). Packs hi halves with one
// v_perm_b32: 3 VALU per pair.
__device__ __forceinline__ uint32_t pack_bf2(float lo, float hi) {
    uint32_t a = __builtin_bit_cast(uint32_t, lo) + 0x8000u;
    uint32_t b = __builtin_bit_cast(uint32_t, hi) + 0x8000u;
#if __has_builtin(__builtin_amdgcn_perm)
    // result bytes: [3:2] = a[3:2], [1:0] = b[3:2]
    return __builtin_amdgcn_perm(b, a, 0x03020706u);
#else
    return (b & 0xFFFF0000u) | (a >> 16);
#endif
}
__device__ __forceinline__ uint16_t f2bf(float x) {
    return (uint16_t)((__builtin_bit_cast(uint32_t, x) + 0x8000u) >> 16);
}

__device__ __forceinline__ v8f wmma_bf16(v16bf a, v16bf b, v8f c) {
    return __builtin_amdgcn_wmma_f32_16x16x32_bf16(
        false, a, false, b, (short)0, c, false, false);
}

// A fragment: 16x32 bf16 tile at (row0, k0) from LDS [rows][stride] (bf16 units).
// lane L: M = row0 + (L&15); elems e -> K = 16*(L>>4)+e
__device__ __forceinline__ v16bf load_a_frag(const uint16_t* lds, int row0, int k0,
                                             int stride, int lane) {
    const uint16_t* base = lds + (row0 + (lane & 15)) * stride + k0;
    int kh = (lane >> 4) * 16;
    union { uint32_t u[8]; v16bf v; } f;
#pragma unroll
    for (int p = 0; p < 8; ++p)
        f.u[p] = *(const uint32_t*)(base + kh + 2 * p);
    return f.v;
}

// B fragment: 32x16 bf16 tile, cols at n0, k at k0, from LDS stored [n][k] (bf16 units).
// lane L: N = n0 + (L&15); elems e -> K = k0 + 16*(L>>4) + e
__device__ __forceinline__ v16bf load_b_frag(const uint16_t* lds, int n0, int k0,
                                             int stride, int lane) {
    const uint16_t* base = lds + (n0 + (lane & 15)) * stride + k0 + (lane >> 4) * 16;
    union { uint32_t u[8]; v16bf v; } f;
#pragma unroll
    for (int p = 0; p < 8; ++p)
        f.u[p] = *(const uint32_t*)(base + 2 * p);
    return f.v;
}

__device__ __forceinline__ float gelu_exact(float x) {
    return 0.5f * x * (1.0f + erff(x * 0.70710678118654752f));
}

__device__ __forceinline__ void store_x_chunk(uint16_t* buf, int r, int c0,
                                              float4 x0, float4 x1) {
    uint32_t* d = (uint32_t*)&buf[r * 34 + c0];
    d[0] = pack_bf2(x0.x, x0.y);
    d[1] = pack_bf2(x0.z, x0.w);
    d[2] = pack_bf2(x1.x, x1.y);
    d[3] = pack_bf2(x1.z, x1.w);
}
__device__ __forceinline__ void store_w_chunk(uint16_t* buf, int kk, int n0,
                                              float4 a, float4 c) {
    uint32_t p0 = pack_bf2(a.x, a.y);
    uint32_t p1 = pack_bf2(a.z, a.w);
    uint32_t p2 = pack_bf2(c.x, c.y);
    uint32_t p3 = pack_bf2(c.z, c.w);
    // low/high halves map to ds_store_b16 / ds_store_b16_d16_hi (no extra VALU)
    buf[(n0 + 0) * 34 + kk] = (uint16_t)p0;
    buf[(n0 + 1) * 34 + kk] = (uint16_t)(p0 >> 16);
    buf[(n0 + 2) * 34 + kk] = (uint16_t)p1;
    buf[(n0 + 3) * 34 + kk] = (uint16_t)(p1 >> 16);
    buf[(n0 + 4) * 34 + kk] = (uint16_t)p2;
    buf[(n0 + 5) * 34 + kk] = (uint16_t)(p2 >> 16);
    buf[(n0 + 6) * 34 + kk] = (uint16_t)p3;
    buf[(n0 + 7) * 34 + kk] = (uint16_t)(p3 >> 16);
}

// ---------------------------------------------------------------------------
// Kernel 1: scores[b,h,s,o] = GELU(X W1 + b1) W2 + b2   (stored fp16 in ws)
// grid.x = B*(S/64)*H (h fastest -> head-blocks sharing an X tile are adjacent
// and share the 256 MiB features stream through the 192 MB L2)
// block = 256 (8 waves). Each WG: 64 seq rows x one head.
// K-loop is double-buffered: next chunk's global loads issue before the
// current chunk's WMMAs so HBM/L2 latency overlaps matrix math; 1 barrier/iter.
// ---------------------------------------------------------------------------
__global__ __launch_bounds__(256) void atnpool_scores_kernel(
    const float* __restrict__ feat,   // [B,S,D]
    const float* __restrict__ w1,     // [H,D,DHEAD]
    const float* __restrict__ b1,     // [H,DHEAD]
    const float* __restrict__ w2,     // [H,DHEAD,DHOUT]
    const float* __restrict__ b2,     // [H,DHOUT]
    _Float16* __restrict__ scores)    // [B,H,S,DHOUT]
{
    __shared__ uint16_t lds_xc[2][64 * 34];  // X chunks  [64 rows][32 k] bf16, pad 2
    __shared__ uint16_t lds_wc[2][64 * 34];  // W1 chunks [64 n][32 k] (transposed), pad 2
    __shared__ uint16_t lds_h [64 * 66];     // hidden    [64 rows][64 k] bf16, pad 2
    __shared__ uint16_t lds_w2[128 * 66];    // W2        [128 n][64 k] (transposed), pad 2

    const int h    = blockIdx.x & (NHEAD - 1);
    const int tile = blockIdx.x >> 3;
    const int b    = tile >> 5;             // S/64 == 32
    const int s0   = (tile & 31) * 64;

    const int t    = threadIdx.x;
    const int lane = t & 31;
    const int wv   = t >> 5;
    const int mi   = wv & 3;                // M tile index (16 rows each)
    const int nset = wv >> 2;               // N half (phase A) / N quad (phase B)

    // per-thread staging coordinates
    const int xr  = t >> 2;                 // X row       (4 threads/row)
    const int xc0 = (t & 3) * 8;            // X col start
    const int wk  = t >> 3;                 // W1 k row    (8 threads/row)
    const int wn0 = (t & 7) * 8;            // W1 n start

    const float* xbase  = feat + ((size_t)b * SEQ + s0) * DIM;
    const float* w1base = w1 + (size_t)h * DIM * DHEAD;

    // ---- stage W2[h] transposed into LDS (bf16) : [k][n] -> [n][k] ----
    {
        const float* w2base = w2 + (size_t)h * DHEAD * DHOUT;
#pragma unroll 4
        for (int i = t; i < DHEAD * DHOUT; i += 256) {
            int k = i >> 7;           // /128
            int n = i & 127;
            lds_w2[n * 66 + k] = f2bf(w2base[i]);
        }
    }

    // ---- prologue: chunk 0 -> LDS buffer 0 ----
    {
        const float4* px = (const float4*)(xbase + (size_t)xr * DIM + xc0);
        float4 x0 = px[0], x1 = px[1];
        const float4* pw = (const float4*)(w1base + (size_t)wk * DHEAD + wn0);
        float4 wA = pw[0], wB = pw[1];
        store_x_chunk(lds_xc[0], xr, xc0, x0, x1);
        store_w_chunk(lds_wc[0], wk, wn0, wA, wB);
    }
    __syncthreads();

    // ---- Phase A: H = X @ W1[h], K = 1024 in chunks of 32, double-buffered ----
    v8f acc0 = {}, acc1 = {};
    for (int kc = 0; kc < DIM / 32; ++kc) {
        const int cur  = kc & 1;
        const bool more = (kc + 1) < (DIM / 32);

        float4 x0, x1, wA, wB;
        if (more) {  // issue next chunk's global loads before this chunk's math
            const float4* px =
                (const float4*)(xbase + (size_t)xr * DIM + (kc + 1) * 32 + xc0);
            x0 = px[0];
            x1 = px[1];
            const float4* pw =
                (const float4*)(w1base + (size_t)((kc + 1) * 32 + wk) * DHEAD + wn0);
            wA = pw[0];
            wB = pw[1];
        }

        v16bf a   = load_a_frag(lds_xc[cur], mi * 16, 0, 34, lane);
        v16bf bf0 = load_b_frag(lds_wc[cur], nset * 32,      0, 34, lane);
        v16bf bf1 = load_b_frag(lds_wc[cur], nset * 32 + 16, 0, 34, lane);
        acc0 = wmma_bf16(a, bf0, acc0);
        acc1 = wmma_bf16(a, bf1, acc1);

        if (more) {  // convert + park next chunk in the alternate buffer
            store_x_chunk(lds_xc[cur ^ 1], xr, xc0, x0, x1);
            store_w_chunk(lds_wc[cur ^ 1], wk, wn0, wA, wB);
        }
        __syncthreads();
    }

    // ---- bias + exact GELU, write hidden tile (bf16) to LDS ----
    {
        int ncol0 = nset * 32 + (lane & 15);
        int ncol1 = ncol0 + 16;
        float bias0 = b1[h * DHEAD + ncol0];
        float bias1 = b1[h * DHEAD + ncol1];
        int rbase = mi * 16 + ((lane >> 4) << 3);
#pragma unroll
        for (int v = 0; v < 8; ++v) {
            int row = rbase + v;
            lds_h[row * 66 + ncol0] = f2bf(gelu_exact(acc0[v] + bias0));
            lds_h[row * 66 + ncol1] = f2bf(gelu_exact(acc1[v] + bias1));
        }
    }
    __syncthreads();

    // ---- Phase B: scores tile [64 x 128] = Htile[64x64] @ W2[64x128] ----
    v8f accB[4] = {};
#pragma unroll
    for (int kc2 = 0; kc2 < 2; ++kc2) {
        v16bf a = load_a_frag(lds_h, mi * 16, kc2 * 32, 66, lane);
#pragma unroll
        for (int j = 0; j < 4; ++j) {
            v16bf bb = load_b_frag(lds_w2, (nset * 4 + j) * 16, kc2 * 32, 66, lane);
            accB[j] = wmma_bf16(a, bb, accB[j]);
        }
    }

    // ---- add b2, store scores fp16: scores[((b*H+h)*S + s)*128 + n] ----
    {
        _Float16* sdst = scores + ((size_t)(b * NHEAD + h) * SEQ + s0) * DHOUT;
        int rbase = mi * 16 + ((lane >> 4) << 3);
#pragma unroll
        for (int j = 0; j < 4; ++j) {
            int n = (nset * 4 + j) * 16 + (lane & 15);
            float bias = b2[h * DHOUT + n];
#pragma unroll
            for (int v = 0; v < 8; ++v) {
                int row = rbase + v;
                sdst[(size_t)row * DHOUT + n] = (_Float16)(accB[j][v] + bias);
            }
        }
    }
}

// ---------------------------------------------------------------------------
// Kernel 2: fused online softmax over S + pooling.
// out[b, h*128+o] = sum_s softmax_s(scores[b,h,s,o]) * feat[b,s,h*128+o]
// grid = B*H blocks, 256 threads: o = t&127, two S-partitions merged in LDS.
// ---------------------------------------------------------------------------
__global__ __launch_bounds__(256) void atnpool_softmax_pool_kernel(
    const float* __restrict__ feat,
    const int* __restrict__ mask,
    const _Float16* __restrict__ scores,
    float* __restrict__ out)
{
    const int bh = blockIdx.x;
    const int b  = bh >> 3;
    const int h  = bh & (NHEAD - 1);
    const int t  = threadIdx.x;
    const int o  = t & (DHOUT - 1);
    const int part = t >> 7;                 // 0 or 1

    const _Float16* sc = scores + (size_t)bh * SEQ * DHOUT + o;
    const float*    fp = feat + (size_t)b * SEQ * DIM + h * DHOUT + o;
    const int*      mp = mask + (size_t)b * SEQ;

    float m = -1e30f, l = 0.0f, acc = 0.0f;
    const int sbeg = part * (SEQ / 2);
    const int send = sbeg + (SEQ / 2);
    for (int s = sbeg; s < send; ++s) {
        if (mp[s] != 0) {
            float x = (float)sc[(size_t)s * DHOUT];
            float f = fp[(size_t)s * DIM];
            if (x > m) {
                float c = expf(m - x);
                l   = l * c + 1.0f;
                acc = acc * c + f;
                m   = x;
            } else {
                float e = expf(x - m);
                l   += e;
                acc += e * f;
            }
        }
    }

    __shared__ float sm[256], sl[256], sa[256];
    sm[t] = m; sl[t] = l; sa[t] = acc;
    __syncthreads();
    if (part == 0) {
        float m2 = sm[t + 128], l2 = sl[t + 128], a2 = sa[t + 128];
        float M  = fmaxf(m, m2);
        float c1 = expf(m - M);
        float c2 = expf(m2 - M);
        float L  = l * c1 + l2 * c2;
        float A  = acc * c1 + a2 * c2;
        out[(size_t)b * DIM + h * DHOUT + o] = (L > 0.0f) ? (A / L) : 0.0f;
    }
}

extern "C" void kernel_launch(void* const* d_in, const int* in_sizes, int n_in,
                              void* d_out, int out_size, void* d_ws, size_t ws_size,
                              hipStream_t stream) {
    const float* feat = (const float*)d_in[0];
    const int*   mask = (const int*)d_in[1];
    // d_in[2] = lengths (unused by the reference forward)
    const float* w1 = (const float*)d_in[3];
    const float* b1 = (const float*)d_in[4];
    const float* w2 = (const float*)d_in[5];
    const float* b2 = (const float*)d_in[6];
    float* out = (float*)d_out;

    _Float16* scores = (_Float16*)d_ws;     // B*H*S*DHOUT fp16 = 128 MiB

    // h fastest in grid.x so head-blocks of the same X tile share L2
    dim3 g1(BATCH * (SEQ / 64) * NHEAD);    // 8192
    atnpool_scores_kernel<<<g1, 256, 0, stream>>>(feat, w1, b1, w2, b2, scores);

    dim3 g2(BATCH * NHEAD);                 // 256
    atnpool_softmax_pool_kernel<<<g2, 256, 0, stream>>>(feat, mask, scores, out);
}
